// DecLayer_2448131358771
// MI455X (gfx1250) — compile-verified, matrix operand order checked by reading, not run on hardware
//
#include <hip/hip_runtime.h>

// ---------------- types & helpers ----------------
typedef __attribute__((ext_vector_type(8)))  __bf16 v8bf;
typedef __attribute__((ext_vector_type(16))) __bf16 v16bf;
typedef __attribute__((ext_vector_type(8)))  float  v8f;

__device__ __forceinline__ unsigned short f2bf(float f) {
    unsigned u = __float_as_uint(f);
    unsigned r = u + 0x7FFFu + ((u >> 16) & 1u);   // round-to-nearest-even
    return (unsigned short)(r >> 16);
}
__device__ __forceinline__ float bf2f(unsigned short s) {
    return __uint_as_float(((unsigned)s) << 16);
}
// branch-free gelu (tanh form): tanh(|u|) = (1-e)/(1+e), e = exp(-2|u|)
__device__ __forceinline__ float gelu_tanh(float x) {
    float u = 0.7978845608028654f * (x + 0.044715f * x * x * x);
    float a = __builtin_fabsf(u);
    float e = __expf(-2.0f * a);                       // v_exp_f32 path
    float t = (1.0f - e) * __builtin_amdgcn_rcpf(1.0f + e);
    t = (u < 0.0f) ? -t : t;
    return 0.5f * x * (1.0f + t);
}

// A fragment: 16x32 bf16 tile from LDS (row-major, `stride` bf16 elems per row).
// Lane<16: row M=lane, K = kt*32 + {0..7, 16..23}; lane>=16: row M=lane-16, K = +8.
__device__ __forceinline__ v16bf load_a(const unsigned short* lds, int stride,
                                        int mt, int kt, int lane) {
    const unsigned short* p = lds + (mt * 16 + (lane & 15)) * stride
                                  + kt * 32 + (lane >> 4) * 8;
    v8bf lo = *(const v8bf*)p;
    v8bf hi = *(const v8bf*)(p + 16);
    return __builtin_shufflevector(lo, hi, 0,1,2,3,4,5,6,7,8,9,10,11,12,13,14,15);
}
// B fragment: pre-swizzled weights, contiguous 16 bf16 per lane per fragment.
__device__ __forceinline__ v16bf load_b(const unsigned short* swz, int ntiles,
                                        int kt, int nt, int lane) {
    const unsigned short* p = swz + (((kt * ntiles + nt) * 32) + lane) * 16;
    return *(const v16bf*)p;
}

__device__ __forceinline__ v8f wmma_bf16(v16bf a, v16bf b, v8f c) {
    return __builtin_amdgcn_wmma_f32_16x16x32_bf16(false, a, false, b,
                                                   (short)0, c, false, false);
}

// ---------------- weight swizzle: f32 [K x (ntiles*16)] -> bf16 fragments ----------------
__global__ void swz_kernel(const float* __restrict__ src, unsigned short* __restrict__ dst,
                           int ktiles, int ntiles) {
    int total = ktiles * ntiles * 512;
    int ncols = ntiles * 16;
    for (int e = blockIdx.x * blockDim.x + threadIdx.x; e < total;
         e += gridDim.x * blockDim.x) {
        int frag = e >> 9;
        int r    = e & 511;
        int lane = r >> 4;
        int i    = r & 15;
        int kt   = frag / ntiles;
        int nt   = frag - kt * ntiles;
        int hi   = lane >> 4;
        int col  = nt * 16 + (lane & 15);
        int k    = kt * 32 + hi * 8 + (i < 8 ? i : i + 8);
        dst[e] = f2bf(src[k * ncols + col]);
    }
}

// ---------------- constants ----------------
#define NB     4
#define NN     2048
#define KK     48
#define HH     128
#define BN_TOT (NB * NN)

// ---------------- K1: node message MLP + reduce + LN1 ----------------
__global__ __launch_bounds__(128) void k1_node_msg(
    const float* __restrict__ hV, const float* __restrict__ hE,
    const float* __restrict__ maskAtt,
    const float* __restrict__ b1, const float* __restrict__ b2,
    const float* __restrict__ b3,
    const float* __restrict__ ln1g, const float* __restrict__ ln1b,
    const unsigned short* __restrict__ w1s,
    const unsigned short* __restrict__ w2s,
    const unsigned short* __restrict__ w3s,
    float* __restrict__ outHV)
{
    __shared__ unsigned short A1[48 * 256];  // 24 KB: concat / reused stage-3 input
    __shared__ unsigned short A2[48 * 128];  // 12 KB: stage-2 input
    __shared__ float hv[128];
    __shared__ float dh[128];
    __shared__ float redS[4], redQ[4];

    const int bn   = blockIdx.x;
    const int tid  = threadIdx.x;
    const int lane = tid & 31;
    const int wave = tid >> 5;

    hv[tid] = hV[bn * 128 + tid];
    dh[tid] = 0.0f;
    __syncthreads();

    for (int idx = tid; idx < 48 * 128; idx += 128) {
        int row = idx >> 7, col = idx & 127;
        A1[row * 256 + col]       = f2bf(hv[col]);
        A1[row * 256 + 128 + col] = f2bf(hE[(bn * 48 + row) * 128 + col]);
    }
    __syncthreads();

    // GEMM1: [48x256] @ W1 + b1 -> gelu -> A2 (stride 128)
    for (int t = wave; t < 24; t += 4) {
        int mt = t >> 3, nt = t & 7;
        v8f c = {};
#pragma unroll
        for (int kt = 0; kt < 8; kt++)
            c = wmma_bf16(load_a(A1, 256, mt, kt, lane), load_b(w1s, 8, kt, nt, lane), c);
        float bias   = b1[nt * 16 + (lane & 15)];
        int colBase  = nt * 16 + (lane & 15);
        int rowBase  = mt * 16 + ((lane >> 4) << 3);
#pragma unroll
        for (int r = 0; r < 8; r++)
            A2[(rowBase + r) * 128 + colBase] = f2bf(gelu_tanh(c[r] + bias));
    }
    __syncthreads();

    // GEMM2: A2 @ W2 + b2 -> gelu -> A1 (reused, stride 128)
    for (int t = wave; t < 24; t += 4) {
        int mt = t >> 3, nt = t & 7;
        v8f c = {};
#pragma unroll
        for (int kt = 0; kt < 4; kt++)
            c = wmma_bf16(load_a(A2, 128, mt, kt, lane), load_b(w2s, 8, kt, nt, lane), c);
        float bias   = b2[nt * 16 + (lane & 15)];
        int colBase  = nt * 16 + (lane & 15);
        int rowBase  = mt * 16 + ((lane >> 4) << 3);
#pragma unroll
        for (int r = 0; r < 8; r++)
            A1[(rowBase + r) * 128 + colBase] = f2bf(gelu_tanh(c[r] + bias));
    }
    __syncthreads();

    // GEMM3: A1 @ W3 + b3, * mask_attend, reduce over 48 edges into dh
    for (int t = wave; t < 24; t += 4) {
        int mt = t >> 3, nt = t & 7;
        v8f c = {};
#pragma unroll
        for (int kt = 0; kt < 4; kt++)
            c = wmma_bf16(load_a(A1, 128, mt, kt, lane), load_b(w3s, 8, kt, nt, lane), c);
        float bias   = b3[nt * 16 + (lane & 15)];
        int colBase  = nt * 16 + (lane & 15);
        int rowBase  = mt * 16 + ((lane >> 4) << 3);
#pragma unroll
        for (int r = 0; r < 8; r++) {
            int row = rowBase + r;
            float m = maskAtt[bn * 48 + row];
            atomicAdd(&dh[colBase], (c[r] + bias) * m);
        }
    }
    __syncthreads();

    // LN1 over 128 features (block-wide reduce)
    float x = hv[tid] + dh[tid] * (1.0f / 30.0f);
    float s = x, q = x * x;
#pragma unroll
    for (int o = 16; o >= 1; o >>= 1) {
        s += __shfl_xor(s, o, 32);
        q += __shfl_xor(q, o, 32);
    }
    if (lane == 0) { redS[wave] = s; redQ[wave] = q; }
    __syncthreads();
    float S  = redS[0] + redS[1] + redS[2] + redS[3];
    float Q  = redQ[0] + redQ[1] + redQ[2] + redQ[3];
    float mu = S * (1.0f / 128.0f);
    float vr = Q * (1.0f / 128.0f) - mu * mu;
    float iv = rsqrtf(vr + 1e-5f);
    outHV[bn * 128 + tid] = (x - mu) * iv * ln1g[tid] + ln1b[tid];
}

// ---------------- K2: FFN (16 tokens/WG) + LN2 + mask_V ----------------
__global__ __launch_bounds__(128) void k2_ffn(
    const float* __restrict__ maskV,
    const float* __restrict__ binb, const float* __restrict__ boutb,
    const float* __restrict__ ln2g, const float* __restrict__ ln2b,
    const unsigned short* __restrict__ wins,
    const unsigned short* __restrict__ wouts,
    float* __restrict__ outHV)
{
    __shared__ unsigned short X[16 * 128];    // 4 KB
    __shared__ unsigned short H1[16 * 512];   // 16 KB
    __shared__ float xres[16 * 128];          // 8 KB
    __shared__ float h2[16 * 128];            // 8 KB

    const int base = blockIdx.x * 16;
    const int tid  = threadIdx.x;
    const int lane = tid & 31;
    const int wave = tid >> 5;

    for (int idx = tid; idx < 16 * 128; idx += 128) {
        int row = idx >> 7, col = idx & 127;
        float v = outHV[(base + row) * 128 + col];
        xres[idx] = v;
        X[row * 128 + col] = f2bf(v);
    }
    __syncthreads();

    // h1 = gelu(X @ Win + bin): [16x128]@[128x512]
    for (int nt = wave; nt < 32; nt += 4) {
        v8f c = {};
#pragma unroll
        for (int kt = 0; kt < 4; kt++)
            c = wmma_bf16(load_a(X, 128, 0, kt, lane), load_b(wins, 32, kt, nt, lane), c);
        float bias  = binb[nt * 16 + (lane & 15)];
        int colBase = nt * 16 + (lane & 15);
        int rowBase = (lane >> 4) << 3;
#pragma unroll
        for (int r = 0; r < 8; r++)
            H1[(rowBase + r) * 512 + colBase] = f2bf(gelu_tanh(c[r] + bias));
    }
    __syncthreads();

    // h2 = h1 @ Wout + bout: [16x512]@[512x128]
    for (int nt = wave * 2; nt < wave * 2 + 2; nt++) {
        v8f c = {};
#pragma unroll
        for (int kt = 0; kt < 16; kt++)
            c = wmma_bf16(load_a(H1, 512, 0, kt, lane), load_b(wouts, 8, kt, nt, lane), c);
        float bias  = boutb[nt * 16 + (lane & 15)];
        int colBase = nt * 16 + (lane & 15);
        int rowBase = (lane >> 4) << 3;
#pragma unroll
        for (int r = 0; r < 8; r++)
            h2[(rowBase + r) * 128 + colBase] = c[r] + bias;
    }
    __syncthreads();

    // residual + LN2 + mask_V; one wave per row group
    for (int row = wave; row < 16; row += 4) {
        float xv[4];
        float s = 0.0f, q = 0.0f;
#pragma unroll
        for (int j = 0; j < 4; j++) {
            int col = lane + 32 * j;
            float v = xres[row * 128 + col] + h2[row * 128 + col];
            xv[j] = v; s += v; q += v * v;
        }
#pragma unroll
        for (int o = 16; o >= 1; o >>= 1) {
            s += __shfl_xor(s, o, 32);
            q += __shfl_xor(q, o, 32);
        }
        float mu = s * (1.0f / 128.0f);
        float vr = q * (1.0f / 128.0f) - mu * mu;
        float iv = rsqrtf(vr + 1e-5f);
        float mk = maskV[base + row];
#pragma unroll
        for (int j = 0; j < 4; j++) {
            int col = lane + 32 * j;
            outHV[(base + row) * 128 + col] =
                ((xv[j] - mu) * iv * ln2g[col] + ln2b[col]) * mk;
        }
    }
}

// ---------------- K3: edge update MLP + LN3 ----------------
__global__ __launch_bounds__(128) void k3_edge(
    const float* __restrict__ hE, const int* __restrict__ Eidx,
    const float* __restrict__ b11, const float* __restrict__ b12,
    const float* __restrict__ b13,
    const float* __restrict__ ln3g, const float* __restrict__ ln3b,
    const unsigned short* __restrict__ w11s,
    const unsigned short* __restrict__ w12s,
    const unsigned short* __restrict__ w13s,
    const float* __restrict__ hVfin, float* __restrict__ outHE)
{
    __shared__ unsigned short A1[48 * 384];  // 36 KB: concat / reused stage-3 input
    __shared__ unsigned short A2[48 * 128];  // 12 KB: stage-2 input, then bf16 msg
    __shared__ int nbr[48];
    __shared__ float cen[128];

    const int bn   = blockIdx.x;
    const int b    = bn >> 11;               // N = 2048
    const int tid  = threadIdx.x;
    const int lane = tid & 31;
    const int wave = tid >> 5;

    if (tid < 48) nbr[tid] = Eidx[bn * 48 + tid];
    cen[tid] = hVfin[bn * 128 + tid];
    __syncthreads();

    for (int idx = tid; idx < 48 * 128; idx += 128) {
        int row = idx >> 7, col = idx & 127;
        A1[row * 384 + col]       = f2bf(cen[col]);
        A1[row * 384 + 128 + col] = f2bf(hE[(bn * 48 + row) * 128 + col]);
        A1[row * 384 + 256 + col] = f2bf(hVfin[(b * NN + nbr[row]) * 128 + col]);
    }
    __syncthreads();

    // GEMM1: [48x384] @ W11 + b11 -> gelu -> A2
    for (int t = wave; t < 24; t += 4) {
        int mt = t >> 3, nt = t & 7;
        v8f c = {};
#pragma unroll
        for (int kt = 0; kt < 12; kt++)
            c = wmma_bf16(load_a(A1, 384, mt, kt, lane), load_b(w11s, 8, kt, nt, lane), c);
        float bias  = b11[nt * 16 + (lane & 15)];
        int colBase = nt * 16 + (lane & 15);
        int rowBase = mt * 16 + ((lane >> 4) << 3);
#pragma unroll
        for (int r = 0; r < 8; r++)
            A2[(rowBase + r) * 128 + colBase] = f2bf(gelu_tanh(c[r] + bias));
    }
    __syncthreads();

    // GEMM2: A2 @ W12 + b12 -> gelu -> A1 (reused, stride 128)
    for (int t = wave; t < 24; t += 4) {
        int mt = t >> 3, nt = t & 7;
        v8f c = {};
#pragma unroll
        for (int kt = 0; kt < 4; kt++)
            c = wmma_bf16(load_a(A2, 128, mt, kt, lane), load_b(w12s, 8, kt, nt, lane), c);
        float bias  = b12[nt * 16 + (lane & 15)];
        int colBase = nt * 16 + (lane & 15);
        int rowBase = mt * 16 + ((lane >> 4) << 3);
#pragma unroll
        for (int r = 0; r < 8; r++)
            A1[(rowBase + r) * 128 + colBase] = f2bf(gelu_tanh(c[r] + bias));
    }
    __syncthreads();

    // GEMM3: A1 @ W13 + b13 -> msg (bf16, into A2)
    for (int t = wave; t < 24; t += 4) {
        int mt = t >> 3, nt = t & 7;
        v8f c = {};
#pragma unroll
        for (int kt = 0; kt < 4; kt++)
            c = wmma_bf16(load_a(A1, 128, mt, kt, lane), load_b(w13s, 8, kt, nt, lane), c);
        float bias  = b13[nt * 16 + (lane & 15)];
        int colBase = nt * 16 + (lane & 15);
        int rowBase = mt * 16 + ((lane >> 4) << 3);
#pragma unroll
        for (int r = 0; r < 8; r++)
            A2[(rowBase + r) * 128 + colBase] = f2bf(c[r] + bias);
    }
    __syncthreads();

    // residual + LN3 per edge row; one wave per row group
    for (int row = wave; row < 48; row += 4) {
        float xv[4];
        float s = 0.0f, q = 0.0f;
#pragma unroll
        for (int j = 0; j < 4; j++) {
            int col = lane + 32 * j;
            float v = hE[(bn * 48 + row) * 128 + col] + bf2f(A2[row * 128 + col]);
            xv[j] = v; s += v; q += v * v;
        }
#pragma unroll
        for (int o = 16; o >= 1; o >>= 1) {
            s += __shfl_xor(s, o, 32);
            q += __shfl_xor(q, o, 32);
        }
        float mu = s * (1.0f / 128.0f);
        float vr = q * (1.0f / 128.0f) - mu * mu;
        float iv = rsqrtf(vr + 1e-5f);
#pragma unroll
        for (int j = 0; j < 4; j++) {
            int col = lane + 32 * j;
            outHE[(bn * 48 + row) * 128 + col] =
                (xv[j] - mu) * iv * ln3g[col] + ln3b[col];
        }
    }
}

// ---------------- host launcher ----------------
extern "C" void kernel_launch(void* const* d_in, const int* in_sizes, int n_in,
                              void* d_out, int out_size, void* d_ws, size_t ws_size,
                              hipStream_t stream) {
    (void)in_sizes; (void)n_in; (void)out_size; (void)ws_size;

    const float* hV      = (const float*)d_in[0];
    const float* hE      = (const float*)d_in[1];
    const float* maskV   = (const float*)d_in[2];
    const int*   Eidx    = (const int*)d_in[3];
    const float* maskAtt = (const float*)d_in[4];
    const float* W1w  = (const float*)d_in[5];  const float* W1b  = (const float*)d_in[6];
    const float* W2w  = (const float*)d_in[7];  const float* W2b  = (const float*)d_in[8];
    const float* W3w  = (const float*)d_in[9];  const float* W3b  = (const float*)d_in[10];
    const float* W11w = (const float*)d_in[11]; const float* W11b = (const float*)d_in[12];
    const float* W12w = (const float*)d_in[13]; const float* W12b = (const float*)d_in[14];
    const float* W13w = (const float*)d_in[15]; const float* W13b = (const float*)d_in[16];
    const float* Winw = (const float*)d_in[17]; const float* Winb = (const float*)d_in[18];
    const float* Woutw= (const float*)d_in[19]; const float* Woutb= (const float*)d_in[20];
    const float* ln1g = (const float*)d_in[21]; const float* ln1b = (const float*)d_in[22];
    const float* ln2g = (const float*)d_in[23]; const float* ln2b = (const float*)d_in[24];
    const float* ln3g = (const float*)d_in[25]; const float* ln3b = (const float*)d_in[26];

    float* outHV = (float*)d_out;
    float* outHE = outHV + (size_t)BN_TOT * HH;

    unsigned short* wsw = (unsigned short*)d_ws;
    // swizzled bf16 weight offsets (in ushort units)
    const size_t OFF_W1   = 0;        // 8x8 tiles   -> 32768
    const size_t OFF_W2   = 32768;    // 4x8         -> 16384
    const size_t OFF_W3   = 49152;    // 4x8         -> 16384
    const size_t OFF_W11  = 65536;    // 12x8        -> 49152
    const size_t OFF_W12  = 114688;   // 4x8         -> 16384
    const size_t OFF_W13  = 131072;   // 4x8         -> 16384
    const size_t OFF_WIN  = 147456;   // 4x32        -> 65536
    const size_t OFF_WOUT = 212992;   // 16x8        -> 65536

    struct { const float* src; size_t off; int kt, nt; } mats[8] = {
        { W1w,  OFF_W1,   8,  8 }, { W2w,   OFF_W2,   4,  8 },
        { W3w,  OFF_W3,   4,  8 }, { W11w,  OFF_W11, 12,  8 },
        { W12w, OFF_W12,  4,  8 }, { W13w,  OFF_W13,  4,  8 },
        { Winw, OFF_WIN,  4, 32 }, { Woutw, OFF_WOUT, 16, 8 },
    };
    for (int i = 0; i < 8; i++) {
        int total = mats[i].kt * mats[i].nt * 512;
        int grid  = (total + 255) / 256;
        swz_kernel<<<grid, 256, 0, stream>>>(mats[i].src, wsw + mats[i].off,
                                             mats[i].kt, mats[i].nt);
    }

    k1_node_msg<<<BN_TOT, 128, 0, stream>>>(
        hV, hE, maskAtt, W1b, W2b, W3b, ln1g, ln1b,
        wsw + OFF_W1, wsw + OFF_W2, wsw + OFF_W3, outHV);

    k2_ffn<<<BN_TOT / 16, 128, 0, stream>>>(
        maskV, Winb, Woutb, ln2g, ln2b,
        wsw + OFF_WIN, wsw + OFF_WOUT, outHV);

    k3_edge<<<BN_TOT, 128, 0, stream>>>(
        hE, Eidx, W11b, W12b, W13b, ln3g, ln3b,
        wsw + OFF_W11, wsw + OFF_W12, wsw + OFF_W13, outHV, outHE);
}
